// CNN_3D_1949915152764
// MI455X (gfx1250) — matrix operand order repaired and checked by reading, not compiled, not run on hardware
//
#include <hip/hip_runtime.h>
#include <stdint.h>
#include <stddef.h>

// ---------------------------------------------------------------------------
// CDNA5 (gfx1250, wave32) types for WMMA bf16 16x16x32, f32 accumulate
// ---------------------------------------------------------------------------
typedef __attribute__((ext_vector_type(16))) __bf16 v16bf;  // A/B frag: 8 VGPRs
typedef __attribute__((ext_vector_type(8)))  float  v8f;    // C/D frag: 8 VGPRs

struct Pack32 { uint4 lo, hi; };  // 32 bytes == one 16x32 bf16 frag per lane

static __device__ __forceinline__ v16bf make_frag(uint4 lo, uint4 hi) {
    Pack32 p{lo, hi};
    return __builtin_bit_cast(v16bf, p);
}

// ---------------------------------------------------------------------------
// Problem constants (from the reference)
// ---------------------------------------------------------------------------
#define NPTS   60000
#define CIN    20
#define CPAD   32            // channel padding so each tap is one K=32 WMMA
#define GD     200
#define GH     160
#define GW     20

// conv output dims
#define D1 198
#define H1 158
#define W1d 18
#define D2 99
#define H2 79
#define W2d 9
#define D3 50
#define H3 40
#define W3d 5
#define D4 25
#define H4 20
#define W4d 3

#define NV1 (D1*H1*W1d)      // 563112
#define NV2 (D2*H2*W2d)      // 70389
#define NV3 (D3*H3*W3d)      // 10000
#define NV4 (D4*H4*W4d)      // 1500

// d_out float offsets: xyz1, feat1, xyz2, feat2, xyz3, feat3, xyz4, feat4
#define OFF_XYZ1  ((size_t)0)
#define OFF_FEAT1 (OFF_XYZ1  + (size_t)NV1*3)
#define OFF_XYZ2  (OFF_FEAT1 + (size_t)NV1*16)
#define OFF_FEAT2 (OFF_XYZ2  + (size_t)NV2*3)
#define OFF_XYZ3  (OFF_FEAT2 + (size_t)NV2*16)
#define OFF_FEAT3 (OFF_XYZ3  + (size_t)NV3*3)
#define OFF_XYZ4  (OFF_FEAT3 + (size_t)NV3*32)
#define OFF_FEAT4 (OFF_XYZ4  + (size_t)NV4*3)

// ---------------------------------------------------------------------------
// 1) scatter-add: features[N,20] -> dense f32 grid [D][H][W][20]
// ---------------------------------------------------------------------------
__global__ void scatter_kernel(const float* __restrict__ feats,
                               const int* __restrict__ coords,
                               float* __restrict__ dense) {
    int t = blockIdx.x * blockDim.x + threadIdx.x;
    if (t >= NPTS * CIN) return;
    int p = t / CIN, c = t - p * CIN;
    int x = coords[p * 4 + 1];
    int y = coords[p * 4 + 2];
    int z = coords[p * 4 + 3];
    float v = feats[(size_t)p * CIN + c];
    atomicAdd(&dense[(((size_t)x * GH + y) * GW + z) * CIN + c], v);
}

// ---------------------------------------------------------------------------
// 2) pack: dense f32 [vox][20] -> bf16 channels-last [vox][32] (pad pre-zeroed)
// ---------------------------------------------------------------------------
__global__ void pack_bf16_kernel(const float* __restrict__ dense,
                                 uint16_t* __restrict__ out, int nvox) {
    int t = blockIdx.x * blockDim.x + threadIdx.x;
    if (t >= nvox * CIN) return;
    int v = t / CIN, c = t - v * CIN;
    float f = dense[(size_t)v * CIN + c];
    out[(size_t)v * CPAD + c] = __builtin_bit_cast(uint16_t, (__bf16)f);
}

// ---------------------------------------------------------------------------
// 3) weight pre-pack into dense 32x16 bf16 B-fragment layout.
//    frag[(tile*27 + tap)*32 + lane][v]: K=8*half+2v+(v>=4?8:0), +1
//    (table order) ; N = tile*16 + lane%16.  W layout: (O, I, 3,3,3).
// ---------------------------------------------------------------------------
__global__ void prepack_kernel(const float* __restrict__ W, int Cin_real,
                               uint32_t* __restrict__ frag) {
    int bt   = blockIdx.x;            // tile*27 + tap
    int tile = bt / 27;
    int tap  = bt - tile * 27;
    int lane = threadIdx.x;
    int r    = lane & 15;
    int half = lane >> 4;
    int n    = tile * 16 + r;
    uint32_t d[8];
#pragma unroll
    for (int v = 0; v < 8; ++v) {
        int k0 = half * 8 + 2 * v + (v >= 4 ? 8 : 0);
        int k1 = k0 + 1;
        float w0 = (k0 < Cin_real) ? W[((size_t)n * Cin_real + k0) * 27 + tap] : 0.0f;
        float w1 = (k1 < Cin_real) ? W[((size_t)n * Cin_real + k1) * 27 + tap] : 0.0f;
        uint32_t b0 = __builtin_bit_cast(uint16_t, (__bf16)w0);
        uint32_t b1 = __builtin_bit_cast(uint16_t, (__bf16)w1);
        d[v] = b0 | (b1 << 16);
    }
    uint32_t* out = frag + ((size_t)bt * 32 + lane) * 8;
    *(uint4*)(out)     = make_uint4(d[0], d[1], d[2], d[3]);
    *(uint4*)(out + 4) = make_uint4(d[4], d[5], d[6], d[7]);
}

// ---------------------------------------------------------------------------
// Async global -> LDS staging of one N-tile of B fragments (6912 dwords).
// Uses gfx1250 GLOBAL_LOAD_ASYNC_TO_LDS_B128 (ASYNCcnt-tracked) so the data
// never round-trips through VGPRs, then drains ASYNCcnt before the barrier.
// ---------------------------------------------------------------------------
static __device__ __forceinline__ void stage_bfrags_async(
        const uint32_t* __restrict__ src, uint32_t* sB, int tid) {
    uint32_t lds_base = (uint32_t)(size_t)sB;   // low 32 bits = LDS byte offset
    // 6912 dwords == 1728 x b128 chunks, 256 threads
#pragma unroll 2
    for (int c = tid; c < 1728; c += 256) {
        const uint32_t* g = src + (size_t)c * 4;
        uint32_t laddr = lds_base + (uint32_t)c * 16u;
        asm volatile("global_load_async_to_lds_b128 %0, %1, off"
                     :: "v"(laddr), "v"(g) : "memory");
    }
    asm volatile("s_wait_asynccnt 0x0" ::: "memory");
}

// ---------------------------------------------------------------------------
// 4) implicit-GEMM 3x3x3 conv via v_wmma_f32_16x16x32_bf16.
//    All geometry is compile-time so the 27-tap K-loop fully unrolls into
//    27 WMMAs with immediate-offset loads. One wave per 16-voxel M tile.
// ---------------------------------------------------------------------------
template<int DIN, int HIN, int WIN, int DOUT, int HOUT, int WOUT,
         int STRIDE, int PAD, int NTILES>
__global__ __launch_bounds__(256) void conv3d_wmma_kernel(
    const uint16_t* __restrict__ in_raw,
    const uint32_t* __restrict__ bfrag,
    float* __restrict__ feat_out, uint16_t* __restrict__ next_out,
    float* __restrict__ xyz_out,
    const float* __restrict__ voxel_size,
    const float* __restrict__ voxel_offset, float lvl_scale)
{
    __shared__ uint32_t sB[27 * 32 * 8];   // 27,648 B: one N-tile of B frags

    constexpr int Mvox = DOUT * HOUT * WOUT;
    constexpr int Cout = NTILES * 16;

    const __bf16* in = (const __bf16*)in_raw;
    const int tid  = threadIdx.x;
    const int lane = tid & 31;
    const int wid  = tid >> 5;
    const int half = lane >> 4;
    const int r    = lane & 15;
    const int m0   = (blockIdx.x * 8 + wid) * 16;

    // this lane's M row (clamped so loads stay in-bounds; stores are guarded)
    int  m      = m0 + r;
    bool mvalid = (m < Mvox);
    int  mc     = mvalid ? m : (Mvox - 1);
    int  i   = mc / (HOUT * WOUT);
    int  rem = mc - i * (HOUT * WOUT);
    int  j   = rem / WOUT;
    int  k   = rem - j * WOUT;

    // input-space base coordinate for tap (0,0,0)
    const int bi = i * STRIDE - PAD;
    const int bj = j * STRIDE - PAD;
    const int bk = k * STRIDE - PAD;
    // base pointer (may point before the array for PAD>0; only dereferenced
    // when the per-tap bounds test passes)
    const __bf16* abase =
        in + ((ptrdiff_t)((bi * HIN + bj) * WIN + bk)) * CPAD;

    __builtin_prefetch((const void*)abase, 0, 1);   // global_prefetch_b8

    for (int tile = 0; tile < NTILES; ++tile) {
        __syncthreads();
        stage_bfrags_async(bfrag + (size_t)tile * (27 * 32 * 8), sB, tid);
        __syncthreads();

        v8f acc = {0.f, 0.f, 0.f, 0.f, 0.f, 0.f, 0.f, 0.f};

#pragma unroll
        for (int kd = 0; kd < 3; ++kd)
#pragma unroll
        for (int kh = 0; kh < 3; ++kh)
#pragma unroll
        for (int kw = 0; kw < 3; ++kw) {
            constexpr int unused = 0; (void)unused;
            const int tap = (kd * 3 + kh) * 3 + kw;
            const ptrdiff_t toff =
                ((ptrdiff_t)(kd * HIN + kh) * WIN + kw) * CPAD; // compile-time

            uint4 alo, ahi;
            if constexpr (PAD == 0) {
                const uint4* p = (const uint4*)(abase + toff);
                alo = p[half];          // bytes half*16    -> K 0..7 / 8..15
                ahi = p[2 + half];      // bytes 32+half*16 -> K 16..23 / 24..31
            } else {
                bool inb = ((unsigned)(bi + kd) < (unsigned)DIN) &
                           ((unsigned)(bj + kh) < (unsigned)HIN) &
                           ((unsigned)(bk + kw) < (unsigned)WIN);
                alo = make_uint4(0, 0, 0, 0);
                ahi = make_uint4(0, 0, 0, 0);
                if (inb) {
                    const uint4* p = (const uint4*)(abase + toff);
                    alo = p[half];
                    ahi = p[2 + half];
                }
            }
            v16bf a = make_frag(alo, ahi);

            const uint4* bp = (const uint4*)&sB[(tap * 32 + lane) * 8];
            v16bf b = make_frag(bp[0], bp[1]);

            acc = __builtin_amdgcn_wmma_f32_16x16x32_bf16(
                      false, a, false, b, (short)0, acc, false, false);
        }

        // D layout: lanes 0-15 -> M=v, N=lane ; lanes 16-31 -> M=8+v, N=lane-16
        const int n = tile * 16 + r;
        if (m0 + 16 <= Mvox) {          // full tile: unguarded stores
#pragma unroll
            for (int v = 0; v < 8; ++v) {
                int mr = m0 + v + 8 * half;
                float val = acc[v];
                feat_out[(size_t)mr * Cout + n] = val;
                if (next_out)
                    next_out[(size_t)mr * CPAD + n] =
                        __builtin_bit_cast(uint16_t, (__bf16)val);
            }
        } else {
#pragma unroll
            for (int v = 0; v < 8; ++v) {
                int mr = m0 + v + 8 * half;
                if (mr < Mvox) {
                    float val = acc[v];
                    feat_out[(size_t)mr * Cout + n] = val;
                    if (next_out)
                        next_out[(size_t)mr * CPAD + n] =
                            __builtin_bit_cast(uint16_t, (__bf16)val);
                }
            }
        }
    }

    // fused xyz for this level (one lane per voxel row)
    if (half == 0 && mvalid) {
        float sx = voxel_size[0] * lvl_scale;
        float sy = voxel_size[1] * lvl_scale;
        float sz = voxel_size[2] * lvl_scale;
        xyz_out[(size_t)m * 3 + 0] = (float)i * sx + voxel_offset[0];
        xyz_out[(size_t)m * 3 + 1] = (float)j * sy + voxel_offset[1];
        xyz_out[(size_t)m * 3 + 2] = (float)k * sz + voxel_offset[2];
    }
}

// ---------------------------------------------------------------------------
// host
// ---------------------------------------------------------------------------
static inline int ceil_div(long long a, long long b) { return (int)((a + b - 1) / b); }

extern "C" void kernel_launch(void* const* d_in, const int* in_sizes, int n_in,
                              void* d_out, int out_size, void* d_ws, size_t ws_size,
                              hipStream_t stream) {
    (void)in_sizes; (void)n_in; (void)out_size; (void)ws_size;

    const float* feats  = (const float*)d_in[0];
    const int*   coords = (const int*)d_in[1];
    const float* Wc0    = (const float*)d_in[2];
    const float* Wc1    = (const float*)d_in[3];
    const float* Wc2    = (const float*)d_in[4];
    const float* Wc3    = (const float*)d_in[5];
    const float* vsize  = (const float*)d_in[6];
    const float* voff   = (const float*)d_in[7];
    float*       out    = (float*)d_out;

    // workspace layout
    char* ws = (char*)d_ws;
    size_t o = 0;
    float*    dense_f32 = (float*)(ws + o);    o += (size_t)GD*GH*GW*CIN*4;   // 51.2 MB
    uint16_t* dense_bf  = (uint16_t*)(ws + o); o += (size_t)GD*GH*GW*CPAD*2;  // 41.0 MB
    uint16_t* x1        = (uint16_t*)(ws + o); o += (size_t)NV1*CPAD*2;       // 36.0 MB
    uint16_t* x2        = (uint16_t*)(ws + o); o += (size_t)NV2*CPAD*2;       //  4.5 MB
    uint16_t* x3        = (uint16_t*)(ws + o); o += (size_t)NV3*CPAD*2;       //  0.64 MB
    size_t zero_bytes = o;                                                    // zero up to here
    uint32_t* frag1 = (uint32_t*)(ws + o); o += (size_t)1*27*32*8*4;
    uint32_t* frag2 = (uint32_t*)(ws + o); o += (size_t)1*27*32*8*4;
    uint32_t* frag3 = (uint32_t*)(ws + o); o += (size_t)2*27*32*8*4;
    uint32_t* frag4 = (uint32_t*)(ws + o); o += (size_t)4*27*32*8*4;

    hipMemsetAsync(d_ws, 0, zero_bytes, stream);

    scatter_kernel<<<ceil_div((long long)NPTS*CIN, 256), 256, 0, stream>>>(feats, coords, dense_f32);
    pack_bf16_kernel<<<ceil_div((long long)GD*GH*GW*CIN, 256), 256, 0, stream>>>(dense_f32, dense_bf, GD*GH*GW);

    prepack_kernel<<< 1*27, 32, 0, stream>>>(Wc0, CIN, frag1);
    prepack_kernel<<< 1*27, 32, 0, stream>>>(Wc1, 16,  frag2);
    prepack_kernel<<< 2*27, 32, 0, stream>>>(Wc2, 16,  frag3);
    prepack_kernel<<< 4*27, 32, 0, stream>>>(Wc3, 32,  frag4);

    conv3d_wmma_kernel<GD, GH, GW, D1, H1, W1d, 1, 0, 1>
        <<<ceil_div(NV1, 128), 256, 0, stream>>>(
        dense_bf, frag1, out + OFF_FEAT1, x1, out + OFF_XYZ1, vsize, voff, 1.0f);

    conv3d_wmma_kernel<D1, H1, W1d, D2, H2, W2d, 2, 1, 1>
        <<<ceil_div(NV2, 128), 256, 0, stream>>>(
        x1, frag2, out + OFF_FEAT2, x2, out + OFF_XYZ2, vsize, voff, 2.0f);

    conv3d_wmma_kernel<D2, H2, W2d, D3, H3, W3d, 2, 1, 2>
        <<<ceil_div(NV3, 128), 256, 0, stream>>>(
        x2, frag3, out + OFF_FEAT3, x3, out + OFF_XYZ3, vsize, voff, 4.0f);

    conv3d_wmma_kernel<D3, H3, W3d, D4, H4, W4d, 2, 1, 4>
        <<<ceil_div(NV4, 128), 256, 0, stream>>>(
        x3, frag4, out + OFF_FEAT4, (uint16_t*)nullptr, out + OFF_XYZ4, vsize, voff, 8.0f);
}
